// RelativeMultiheadAttentionMechanism_20220706030194
// MI455X (gfx1250) — matrix operand order repaired and checked by reading, not compiled, not run on hardware
//
#include <hip/hip_runtime.h>

// ---------------- problem constants (from reference) ----------------
enum : int {
  cQ  = 1024,      // QLEN
  cM  = 1024,      // MLEN
  cK  = 1024,      // KLEN
  cT  = 2048,      // MLEN+KLEN
  cA  = 1024,      // ADIM == KDIM == QDIM
  cH  = 16,        // heads
  cD  = 64,        // DK
};
#define NEG_INF_F  (-3.4028234663852886e38f)
#define INV_SCALE  (0.125f)   // 1/sqrt(64)

// ---------------- vector types ----------------
typedef __attribute__((ext_vector_type(16))) __bf16          v16bf;
typedef __attribute__((ext_vector_type(16))) unsigned short  v16u;
typedef __attribute__((ext_vector_type(8)))  unsigned short  v8u;
typedef __attribute__((ext_vector_type(8)))  float           v8f;
typedef __attribute__((ext_vector_type(4)))  float           v4f;

// f32 -> bf16 (round-to-nearest-even), bit-level so we never rely on __bf16 arithmetic
__device__ __forceinline__ unsigned short f2bf(float f) {
  unsigned int u = __builtin_bit_cast(unsigned int, f);
  unsigned int r = u + 0x7FFFu + ((u >> 16) & 1u);
  return (unsigned short)(r >> 16);
}

__device__ __forceinline__ v8f wmma_bf16(v16u a, v16u b, v8f c) {
  union U { v16u u; v16bf b; };
  U ua; ua.u = a;
  U ub; ub.u = b;
  return __builtin_amdgcn_wmma_f32_16x16x32_bf16(
      /*neg_a=*/false, ua.b, /*neg_b=*/false, ub.b,
      /*c_mod=*/(short)0, c, /*reuse_a=*/false, /*reuse_b=*/false);
}

// ---- per-thread staging register block: 16 operand elements ----
template <typename T> struct Stage;

template <> struct Stage<unsigned short> {      // bf16 operand: straight copy
  v8u x0, x1;
  __device__ __forceinline__ void load(const unsigned short* src, bool valid) {
    x0 = (v8u){}; x1 = (v8u){};
    if (valid) { x0 = *(const v8u*)src; x1 = *(const v8u*)(src + 8); }
  }
  __device__ __forceinline__ void store(unsigned short* dst) const {
    *(v8u*)dst = x0; *(v8u*)(dst + 8) = x1;
  }
};

template <> struct Stage<float> {               // f32 operand: convert at LDS store
  v4f f0, f1, f2, f3;
  __device__ __forceinline__ void load(const float* src, bool valid) {
    f0 = (v4f){}; f1 = (v4f){}; f2 = (v4f){}; f3 = (v4f){};
    if (valid) {
      f0 = *(const v4f*)src;       f1 = *(const v4f*)(src + 4);
      f2 = *(const v4f*)(src + 8); f3 = *(const v4f*)(src + 12);
    }
  }
  __device__ __forceinline__ void store(unsigned short* dst) const {
#pragma unroll
    for (int i = 0; i < 4; ++i) {
      dst[i]      = f2bf(f0[i]);
      dst[4 + i]  = f2bf(f1[i]);
      dst[8 + i]  = f2bf(f2[i]);
      dst[12 + i] = f2bf(f3[i]);
    }
  }
};

// A-fragment (16x32 bf16, ISA 7.12.2 "16-bit A-Matrix"):
// lanes 0-15 : elems 0-7 -> K 0..7,  elems 8-15 -> K 16..23
// lanes 16-31: elems 0-7 -> K 8..15, elems 8-15 -> K 24..31
__device__ __forceinline__ v16u ldfragA(const unsigned short* rowPtr, int lane) {
  const int kb = (lane & 16) ? 8 : 0;
  v8u lo = *(const v8u*)(rowPtr + kb);
  v8u hi = *(const v8u*)(rowPtr + kb + 16);
  return __builtin_shufflevector(lo, hi, 0,1,2,3,4,5,6,7,8,9,10,11,12,13,14,15);
}
// B-fragment (32x16 bf16): lanes 0-15 hold K=0..7,16..23, lanes 16-31 hold K=8..15,24..31
__device__ __forceinline__ v16u ldfragB(const unsigned short* rowPtr, int lane) {
  const int kb = (lane & 16) ? 8 : 0;
  v8u lo = *(const v8u*)(rowPtr + kb);
  v8u hi = *(const v8u*)(rowPtr + kb + 16);
  return __builtin_shufflevector(lo, hi, 0,1,2,3,4,5,6,7,8,9,10,11,12,13,14,15);
}

// =====================================================================
// Generic batched WMMA GEMM:  C[m,n] = sum_k A[m,k] * Bt[n,k]  (+bias[n])
// Block tile 128x128, 8 waves arranged 4(M) x 2(N); each wave computes a
// 32x64 sub-tile (2 A-frags x 4 B-frags -> 8 WMMA per 32-wide K chunk,
// only 12 ds_load_b128 per chunk vs 18 for a 16x128 wave tile).
// Double-buffered LDS (one barrier per K-step); next-chunk global loads
// are issued before the compute phase so HBM/L2 latency overlaps WMMA.
// TA: float (converted to bf16 at LDS store) or unsigned short (bf16 bits).
// EPI: 0 = store f32, 1 = store bf16, 2 = TransformerXL rel_shift scatter-add
// =====================================================================
template <typename TA, int EPI>
__global__ __launch_bounds__(256) void wmma_gemm(
    const TA* __restrict__ A, const unsigned short* __restrict__ Bt,
    const float* __restrict__ bias, void* __restrict__ Cvoid,
    int M, int N, int K, int lda, int ldb, int ldc,
    long long aOffB, long long aOffH, long long bOffB, long long bOffH,
    long long cOffB, long long cOffH, int batH) {
  constexpr int LSTR = 40;                     // LDS row stride (bank padding)
  __shared__ unsigned short lA[2][128 * LSTR];
  __shared__ unsigned short lB[2][128 * LSTR];

  const int z  = blockIdx.z;
  const int zb = z / batH;
  const int zh = z % batH;
  A  += zb * aOffB + zh * aOffH;
  Bt += zb * bOffB + zh * bOffH;
  float*          Cf = (float*)Cvoid          + zb * cOffB + zh * cOffH;
  unsigned short* Cb = (unsigned short*)Cvoid + zb * cOffB + zh * cOffH;

  const int m0  = blockIdx.y * 128;
  const int n0  = blockIdx.x * 128;
  const int tid = threadIdx.x;

  const int sRow = tid >> 1;          // 0..127 staged row
  const int sCol = (tid & 1) << 4;    // 0 or 16 within 32-wide K chunk
  const bool aValid = (m0 + sRow) < M;
  const bool bValid = (n0 + sRow) < N;
  const TA* aPtr             = A  + (size_t)(m0 + sRow) * lda + sCol;
  const unsigned short* bPtr = Bt + (size_t)(n0 + sRow) * ldb + sCol;
  unsigned short* const ldsA = &lA[0][sRow * LSTR + sCol];
  unsigned short* const ldsB = &lB[0][sRow * LSTR + sCol];
  constexpr int BUFOFF = 128 * LSTR;  // element offset between the two buffers

  const int lane = tid & 31;
  const int wv   = tid >> 5;          // wave 0..7
  const int wvM  = wv & 3;            // wave row group: 32 rows each
  const int wvN  = wv >> 2;           // wave col group: 64 cols each
  const int lrow = lane & 15;
  const int mAdd = (lane & 16) ? 8 : 0;

  v8f acc[2][4];
#pragma unroll
  for (int mi = 0; mi < 2; ++mi)
#pragma unroll
    for (int ni = 0; ni < 4; ++ni) acc[mi][ni] = (v8f){};

  Stage<TA>             sa;
  Stage<unsigned short> sb;

  // prologue: chunk 0 -> LDS buffer 0
  sa.load(aPtr, aValid);
  sb.load(bPtr, bValid);
  sa.store(ldsA);
  sb.store(ldsB);
  __syncthreads();

  const int ksteps = K >> 5;          // K must be a multiple of 32
  for (int kc = 0; kc < ksteps; ++kc) {
    const int cur = kc & 1;
    const bool more = (kc + 1) < ksteps;

    // issue next-chunk global loads first (latency overlaps compute)
    if (more) {
      const int k1 = (kc + 1) << 5;
      sa.load(aPtr + k1, aValid);
      sb.load(bPtr + k1, bValid);
    }

    // compute from LDS[cur]: 2 A-frags x 4 B-frags -> 8 WMMA
    const unsigned short* curA = &lA[cur][(wvM * 32 + lrow) * LSTR];
    const unsigned short* curB = &lB[cur][(wvN * 64 + lrow) * LSTR];
    v16u af[2];
#pragma unroll
    for (int mi = 0; mi < 2; ++mi) af[mi] = ldfragA(curA + mi * 16 * LSTR, lane);
#pragma unroll
    for (int ni = 0; ni < 4; ++ni) {
      v16u bf = ldfragB(curB + ni * 16 * LSTR, lane);
#pragma unroll
      for (int mi = 0; mi < 2; ++mi)
        acc[mi][ni] = wmma_bf16(af[mi], bf, acc[mi][ni]);
    }

    // stage next chunk into the other buffer
    if (more) {
      const int nxt = (cur ^ 1) * BUFOFF;
      sa.store(ldsA + nxt);
      sb.store(ldsB + nxt);
    }
    __syncthreads();
  }

  // epilogue: VGPR r of D holds M = base + r (+8 for lanes 16-31), N = lane&15
#pragma unroll
  for (int mi = 0; mi < 2; ++mi) {
    const int mB = m0 + wvM * 32 + mi * 16 + mAdd;
#pragma unroll
    for (int ni = 0; ni < 4; ++ni) {
      const int gn   = n0 + wvN * 64 + ni * 16 + lrow;
      const float ba = (EPI != 2 && bias != nullptr && gn < N) ? bias[gn] : 0.0f;
#pragma unroll
      for (int r = 0; r < 8; ++r) {
        const int gm = mB + r;
        if (gm < M && gn < N) {
          float v = acc[mi][ni][r] + ba;
          if constexpr (EPI == 0) {
            Cf[(size_t)gm * ldc + gn] = v;
          } else if constexpr (EPI == 1) {
            Cb[(size_t)gm * ldc + gn] = f2bf(v);
          } else {
            // rel_shift scatter: raw(gm, gn) -> flat f = gm*(T+1)+gn+1
            // shifted slot: i=(f-Q)/T, j=(f-Q)%T  (only if f >= Q); race-free
            const int f = gm * (cT + 1) + gn + 1;
            if (f >= cQ) {
              const int t = f - cQ;
              const int i = t / cT;
              const int j = t - i * cT;
              Cf[(size_t)i * cT + j] += v;
            }
          }
        }
      }
    }
  }
}

// ---------------- elementwise kernels ----------------
__global__ __launch_bounds__(256) void cvt_f32_bf16(const float* __restrict__ in,
                                                    unsigned short* __restrict__ out,
                                                    int n) {
  int i = blockIdx.x * blockDim.x + threadIdx.x;
  if (i < n) out[i] = f2bf(in[i]);
}

// qu = bf16(qproj + u[col]), qv = bf16(qproj + v[col]); col = idx % 1024
__global__ __launch_bounds__(256) void quqv_kernel(const float* __restrict__ qp,
                                                   const float* __restrict__ u,
                                                   const float* __restrict__ v,
                                                   unsigned short* __restrict__ qu,
                                                   unsigned short* __restrict__ qv,
                                                   int n) {
  int i = blockIdx.x * blockDim.x + threadIdx.x;
  if (i < n) {
    int col = i & (cA - 1);
    float q = qp[i];
    qu[i] = f2bf(q + u[col]);
    qv[i] = f2bf(q + v[col]);
  }
}

// vT[((b*H+h)*DK+d)*T + j] = vproj[(b*T + j)*A + h*DK + d]
__global__ __launch_bounds__(256) void transpose_v(const unsigned short* __restrict__ vp,
                                                   unsigned short* __restrict__ vt,
                                                   int n) {
  int i = blockIdx.x * blockDim.x + threadIdx.x;
  if (i < n) {
    int j = i & (cT - 1);
    int d = (i >> 11) & (cD - 1);        // /cT % cD
    int h = (i >> 17) & (cH - 1);        // /(cT*cD) % cH
    int b = i >> 21;                     // /(cT*cD*cH)
    vt[i] = vp[((size_t)(b * cT + j)) * cA + h * cD + d];
  }
}

// in-place masked, scaled softmax over last dim (T=2048); one block per row
__global__ __launch_bounds__(256) void softmax_mask(float* __restrict__ e,
                                                    const int* __restrict__ mask) {
  const int row = blockIdx.x;               // b*H*Q + h*Q + i
  const int b   = row / (cH * cQ);
  const int i   = row & (cQ - 1);
  float* er       = e + (size_t)row * cT;
  const int* mr   = mask + ((size_t)b * cQ + i) * cT;
  __shared__ float red[256];
  const int tid = threadIdx.x;

  float vals[8];
  float mx = NEG_INF_F;
#pragma unroll
  for (int t = 0; t < 8; ++t) {
    const int j = tid + t * 256;
    float v = (mr[j] == 0) ? NEG_INF_F : er[j] * INV_SCALE;
    vals[t] = v;
    mx = fmaxf(mx, v);
  }
  red[tid] = mx; __syncthreads();
  for (int s = 128; s > 0; s >>= 1) {
    if (tid < s) red[tid] = fmaxf(red[tid], red[tid + s]);
    __syncthreads();
  }
  mx = red[0]; __syncthreads();

  float sum = 0.0f;
#pragma unroll
  for (int t = 0; t < 8; ++t) { vals[t] = __expf(vals[t] - mx); sum += vals[t]; }
  red[tid] = sum; __syncthreads();
  for (int s = 128; s > 0; s >>= 1) {
    if (tid < s) red[tid] += red[tid + s];
    __syncthreads();
  }
  const float inv = 1.0f / red[0];
#pragma unroll
  for (int t = 0; t < 8; ++t) er[tid + t * 256] = vals[t] * inv;
}

// =====================================================================
extern "C" void kernel_launch(void* const* d_in, const int* in_sizes, int n_in,
                              void* d_out, int out_size, void* d_ws, size_t ws_size,
                              hipStream_t stream) {
  (void)n_in; (void)out_size; (void)ws_size;
  const float* key    = (const float*)d_in[0];
  const float* query  = (const float*)d_in[1];
  const float* memory = (const float*)d_in[2];
  const float* pos    = (const float*)d_in[3];
  const int*   mask   = (const int*)  d_in[4];
  const float* u      = (const float*)d_in[5];
  const float* v      = (const float*)d_in[6];
  const float* Wk = (const float*)d_in[7];  const float* bk = (const float*)d_in[8];
  const float* Wv = (const float*)d_in[9];  const float* bv = (const float*)d_in[10];
  const float* Wq = (const float*)d_in[11]; const float* bq = (const float*)d_in[12];
  const float* Wp = (const float*)d_in[13]; const float* bp = (const float*)d_in[14];
  const float* Wo = (const float*)d_in[15]; const float* bo = (const float*)d_in[16];

  const int Bn = in_sizes[1] / (cQ * cA);   // batch from query size

  float* out_cv = (float*)d_out;
  float* e      = out_cv + (size_t)Bn * cQ * cA;   // logits -> softmax -> aw, in d_out

  // ---- workspace carve-out (bf16 buffers stored as unsigned short) ----
  char* w = (char*)d_ws;
  auto alloc = [&](size_t bytes) -> char* {
    char* p = w; w += (bytes + 255) & ~(size_t)255; return p;
  };
  unsigned short* kcat  = (unsigned short*)alloc((size_t)Bn * cT * cA * 2);
  unsigned short* qb    = (unsigned short*)alloc((size_t)Bn * cQ * cA * 2);
  unsigned short* pb    = (unsigned short*)alloc((size_t)cT * cA * 2);
  unsigned short* wkb   = (unsigned short*)alloc((size_t)cA * cA * 2);
  unsigned short* wvb   = (unsigned short*)alloc((size_t)cA * cA * 2);
  unsigned short* wqb   = (unsigned short*)alloc((size_t)cA * cA * 2);
  unsigned short* wpb   = (unsigned short*)alloc((size_t)cA * cA * 2);
  unsigned short* wob   = (unsigned short*)alloc((size_t)cA * cA * 2);
  unsigned short* kproj = (unsigned short*)alloc((size_t)Bn * cT * cA * 2);
  unsigned short* vproj = (unsigned short*)alloc((size_t)Bn * cT * cA * 2);
  float*          qprojf= (float*)         alloc((size_t)Bn * cQ * cA * 4);
  unsigned short* pproj = (unsigned short*)alloc((size_t)cT * cA * 2);
  unsigned short* qu    = (unsigned short*)alloc((size_t)Bn * cQ * cA * 2);
  unsigned short* qv    = (unsigned short*)alloc((size_t)Bn * cQ * cA * 2);
  unsigned short* vT    = (unsigned short*)alloc((size_t)Bn * cH * cD * cT * 2);
  unsigned short* cvpre = (unsigned short*)alloc((size_t)Bn * cQ * cA * 2);

  const int TPB = 256;
  auto blocks = [](size_t n) { return (unsigned)((n + 255) / 256); };

  // 1) bf16 conversions (kcat = concat(memory, key) per batch)
  for (int b = 0; b < Bn; ++b) {
    cvt_f32_bf16<<<blocks((size_t)cM * cA), TPB, 0, stream>>>(
        memory + (size_t)b * cM * cA, kcat + (size_t)b * cT * cA, cM * cA);
    cvt_f32_bf16<<<blocks((size_t)cK * cA), TPB, 0, stream>>>(
        key + (size_t)b * cK * cA, kcat + (size_t)b * cT * cA + (size_t)cM * cA, cK * cA);
  }
  cvt_f32_bf16<<<blocks((size_t)Bn * cQ * cA), TPB, 0, stream>>>(query, qb, Bn * cQ * cA);
  cvt_f32_bf16<<<blocks((size_t)cT * cA), TPB, 0, stream>>>(pos, pb, cT * cA);
  cvt_f32_bf16<<<blocks((size_t)cA * cA), TPB, 0, stream>>>(Wk, wkb, cA * cA);
  cvt_f32_bf16<<<blocks((size_t)cA * cA), TPB, 0, stream>>>(Wv, wvb, cA * cA);
  cvt_f32_bf16<<<blocks((size_t)cA * cA), TPB, 0, stream>>>(Wq, wqb, cA * cA);
  cvt_f32_bf16<<<blocks((size_t)cA * cA), TPB, 0, stream>>>(Wp, wpb, cA * cA);
  cvt_f32_bf16<<<blocks((size_t)cA * cA), TPB, 0, stream>>>(Wo, wob, cA * cA);

  // 2) projections
  {
    dim3 g(cA / 128, (Bn * cT) / 128, 1);
    wmma_gemm<unsigned short, 1><<<g, TPB, 0, stream>>>(
        kcat, wkb, bk, kproj, Bn * cT, cA, cA, cA, cA, cA, 0,0,0,0,0,0, 1);
    wmma_gemm<unsigned short, 1><<<g, TPB, 0, stream>>>(
        kcat, wvb, bv, vproj, Bn * cT, cA, cA, cA, cA, cA, 0,0,0,0,0,0, 1);
  }
  {
    dim3 g(cA / 128, (Bn * cQ) / 128, 1);
    wmma_gemm<unsigned short, 0><<<g, TPB, 0, stream>>>(
        qb, wqb, bq, qprojf, Bn * cQ, cA, cA, cA, cA, cA, 0,0,0,0,0,0, 1);
  }
  {
    dim3 g(cA / 128, cT / 128, 1);
    wmma_gemm<unsigned short, 1><<<g, TPB, 0, stream>>>(
        pb, wpb, bp, pproj, cT, cA, cA, cA, cA, cA, 0,0,0,0,0,0, 1);
  }

  // 3) q+u / q+v (bf16), value transpose
  quqv_kernel<<<blocks((size_t)Bn * cQ * cA), TPB, 0, stream>>>(
      qprojf, u, v, qu, qv, Bn * cQ * cA);
  transpose_v<<<blocks((size_t)Bn * cH * cD * cT), TPB, 0, stream>>>(
      vproj, vT, Bn * cH * cD * cT);

  // 4) AC logits -> e  (per b,h: [Q,DK] x [T,DK]^T)
  {
    dim3 g(cT / 128, cQ / 128, Bn * cH);
    wmma_gemm<unsigned short, 0><<<g, TPB, 0, stream>>>(
        qu, kproj, nullptr, e, cQ, cT, cD, cA, cA, cT,
        (long long)cQ * cA, cD, (long long)cT * cA, cD,
        (long long)cH * cQ * cT, (long long)cQ * cT, cH);
  }
  // 5) BD raw GEMM with rel_shift scatter-add into e
  {
    dim3 g(cT / 128, cQ / 128, Bn * cH);
    wmma_gemm<unsigned short, 2><<<g, TPB, 0, stream>>>(
        qv, pproj, nullptr, e, cQ, cT, cD, cA, cA, cT,
        (long long)cQ * cA, cD, 0LL, cD,
        (long long)cH * cQ * cT, (long long)cQ * cT, cH);
  }

  // 6) scale + mask + softmax in place -> aw (final output region)
  softmax_mask<<<(unsigned)(Bn * cH * cQ), TPB, 0, stream>>>(e, mask);

  // 7) context: cvpre[b,i,h*64+d] = sum_j aw[b,h,i,j] * vT[b,h,d,j]
  {
    dim3 g(1, cQ / 128, Bn * cH);   // N=64 -> one 128-wide tile (bounds-guarded)
    wmma_gemm<float, 1><<<g, TPB, 0, stream>>>(
        e, vT, nullptr, cvpre, cQ, cD, cT, cT, cT, cA,
        (long long)cH * cQ * cT, (long long)cQ * cT,
        (long long)cH * cD * cT, (long long)cD * cT,
        (long long)cQ * cA, cD, cH);
  }

  // 8) output projection: cv = cvpre @ Wo^T + bo  (f32 into d_out)
  {
    dim3 g(cA / 128, (Bn * cQ) / 128, 1);
    wmma_gemm<unsigned short, 0><<<g, TPB, 0, stream>>>(
        cvpre, wob, bo, out_cv, Bn * cQ, cA, cA, cA, cA, cA, 0,0,0,0,0,0, 1);
  }
}
